// lstm2_5282809774260
// MI455X (gfx1250) — matrix operand order repaired and checked by reading, not compiled, not run on hardware
//
#include <hip/hip_runtime.h>
#include <math.h>

typedef __attribute__((ext_vector_type(16))) __bf16 v16bf;
typedef __attribute__((ext_vector_type(8)))  __bf16 v8bf;
typedef __attribute__((ext_vector_type(8)))  float  v8f;

union AFrag { v16bf v; v8bf h2[2]; };

__device__ __forceinline__ unsigned short f2bf(float f) {
  unsigned u = __float_as_uint(f);
  u += 0x7FFFu + ((u >> 16) & 1u);          // round-to-nearest-even
  return (unsigned short)(u >> 16);
}
__device__ __forceinline__ float sigmoidf_(float x) { return 1.0f / (1.0f + __expf(-x)); }

// ---------------- prep kernels ----------------
__global__ void k_f32_to_bf16(const float* __restrict__ in, unsigned short* __restrict__ out, int n) {
  int i = blockIdx.x * blockDim.x + threadIdx.x;
  if (i < n) out[i] = f2bf(in[i]);
}

__global__ void k_init(unsigned short* __restrict__ hbuf, unsigned* __restrict__ bar, int n) {
  int i = blockIdx.x * blockDim.x + threadIdx.x;
  if (i < n) hbuf[i] = 0;
  if (i == 0) *bar = 0u;
}

// ---------------- persistent LSTM kernel ----------------
// Grid: 32 WGs x 256 threads. WG w owns h-columns [16w, 16w+16).
// Wave j owns batch rows [16j, 16j+16). Cell state c lives in VGPRs for all 512 steps.
__global__ void __launch_bounds__(256, 1)
k_lstm(const unsigned short* __restrict__ Wxb,   // bf16 [128][2048]
       const unsigned short* __restrict__ Whb,   // bf16 [512][2048]
       const unsigned short* __restrict__ xb,    // bf16 [128][512][128]
       const float*          __restrict__ bias,  // f32  [2048]
       unsigned short*       __restrict__ hbuf,  // bf16 [2][128][512] ping-pong
       float*                __restrict__ hlast, // f32  [128][512]
       unsigned*             __restrict__ bar)   // global barrier counter
{
  constexpr int T = 512, F = 128, H = 512, G = 2048;
  __shared__ v16bf ldsW[2048];   // 64 KB: Wh slice pre-swizzled into B-fragments

  const int tid  = threadIdx.x;
  const int wave = tid >> 5;
  const int lane = tid & 31;
  const int li   = lane & 15;
  const int hi   = lane >> 4;
  const int n0   = blockIdx.x * 16;     // owned h-column base
  const int col  = n0 + li;             // this lane's h column
  const int mrow = wave * 16 + li;      // A-fragment row for loads

  // Stage Wh slice: chunk c -> (kblk = c>>7, gate = (c>>5)&3, lane = c&31).
  // Each chunk is one lane's 32B B-fragment: Wh[kblk*32 + k(s,lane>>4)][g*512+n0+(lane&15)], table k order.
  for (int c = tid * 8; c < tid * 8 + 8; ++c) {
    const int kblk = c >> 7;
    const int g    = (c >> 5) & 3;
    const int ln   = c & 31;
    v16bf tmp;
    for (int s = 0; s < 16; ++s) {
      const int kk = (s < 8) ? (8 * (ln >> 4) + s) : (16 + 8 * (ln >> 4) + (s - 8));
      tmp[s] = *(const __bf16*)(Whb + (size_t)(kblk * 32 + kk) * G + g * H + n0 + (ln & 15));
    }
    ldsW[c] = tmp;
  }
  __syncthreads();

  // Per-lane gate biases (constant over t); fold FORGET_BIAS into f-gate.
  const float bi  = bias[0 * H + col];
  const float bj  = bias[1 * H + col];
  const float bf_ = bias[2 * H + col] + 1.0f;
  const float bo  = bias[3 * H + col];

  v8f cst = {};   // cell state: rows 16*wave + r + 8*hi, column col

  const unsigned short* xrowbase = xb + (size_t)mrow * T * F;

  for (int t = 0; t < T; ++t) {
    v8f ai = {}, aj = {}, af = {}, ao = {};

    // ---- recurrent contribution: h(t) [128x512] x Wh [512x2048] ----
    const unsigned short* hrow = hbuf + (t & 1) * (128 * H) + (size_t)mrow * H;
    for (int k0 = 0; k0 < H; k0 += 32) {
      AFrag a;
      a.h2[0] = *(const v8bf*)(hrow + k0 + 8 * hi);        // K = k0 + 8*hi + e
      a.h2[1] = *(const v8bf*)(hrow + k0 + 16 + 8 * hi);   // K = k0 + 16 + 8*hi + e
      const int kb = (k0 >> 5) * 128 + lane;
      v16bf b0 = ldsW[kb];
      v16bf b1 = ldsW[kb + 32];
      v16bf b2 = ldsW[kb + 64];
      v16bf b3 = ldsW[kb + 96];
      ai = __builtin_amdgcn_wmma_f32_16x16x32_bf16(false, a.v, false, b0, (short)0, ai, false, false);
      aj = __builtin_amdgcn_wmma_f32_16x16x32_bf16(false, a.v, false, b1, (short)0, aj, false, false);
      af = __builtin_amdgcn_wmma_f32_16x16x32_bf16(false, a.v, false, b2, (short)0, af, false, false);
      ao = __builtin_amdgcn_wmma_f32_16x16x32_bf16(false, a.v, false, b3, (short)0, ao, false, false);
    }

    // ---- input contribution: x_t [128x128] x Wx [128x2048] (Wx hot in L2) ----
    const unsigned short* xrow = xrowbase + (size_t)t * F;
    for (int k0 = 0; k0 < F; k0 += 32) {
      AFrag a;
      a.h2[0] = *(const v8bf*)(xrow + k0 + 8 * hi);
      a.h2[1] = *(const v8bf*)(xrow + k0 + 16 + 8 * hi);
      v16bf b0, b1, b2, b3;
      for (int s = 0; s < 16; ++s) {
        const int kk = (s < 8) ? (8 * hi + s) : (16 + 8 * hi + (s - 8));
        const unsigned short* wrow = Wxb + (size_t)(k0 + kk) * G + col;
        b0[s] = *(const __bf16*)(wrow + 0 * H);
        b1[s] = *(const __bf16*)(wrow + 1 * H);
        b2[s] = *(const __bf16*)(wrow + 2 * H);
        b3[s] = *(const __bf16*)(wrow + 3 * H);
      }
      ai = __builtin_amdgcn_wmma_f32_16x16x32_bf16(false, a.v, false, b0, (short)0, ai, false, false);
      aj = __builtin_amdgcn_wmma_f32_16x16x32_bf16(false, a.v, false, b1, (short)0, aj, false, false);
      af = __builtin_amdgcn_wmma_f32_16x16x32_bf16(false, a.v, false, b2, (short)0, af, false, false);
      ao = __builtin_amdgcn_wmma_f32_16x16x32_bf16(false, a.v, false, b3, (short)0, ao, false, false);
    }

    // ---- gates + state update (fp32), publish h(t+1) as bf16 ----
    unsigned short* hnext = hbuf + ((t + 1) & 1) * (128 * H);
#pragma unroll
    for (int r = 0; r < 8; ++r) {
      const int m = wave * 16 + r + 8 * hi;   // C/D layout: VGPR r -> row r + 8*hi
      const float zi = ai[r] + bi;
      const float zj = aj[r] + bj;
      const float zf = af[r] + bf_;
      const float zo = ao[r] + bo;
      const float cn = sigmoidf_(zf) * cst[r] + sigmoidf_(zi) * tanhf(zj);
      const float hh = sigmoidf_(zo) * tanhf(cn);
      cst[r] = cn;
      hnext[(size_t)m * H + col] = f2bf(hh);
      if (t == T - 1) hlast[(size_t)m * H + col] = hh;
    }

    // ---- device-wide barrier between timesteps ----
    __builtin_amdgcn_fence(__ATOMIC_RELEASE, "agent");   // each wave flushes its h stores
    __syncthreads();
    if (tid == 0) {
      atomicAdd(bar, 1u);
      const unsigned target = (unsigned)(t + 1) * gridDim.x;
      while (__hip_atomic_load(bar, __ATOMIC_RELAXED, __HIP_MEMORY_SCOPE_AGENT) < target)
        __builtin_amdgcn_s_sleep(2);
    }
    __syncthreads();
    __builtin_amdgcn_fence(__ATOMIC_ACQUIRE, "agent");   // invalidate stale h lines
  }
}

// ---------------- head: BN + dense + softmax + per-row loss ----------------
__global__ void k_head(const float* __restrict__ hlast, const float* __restrict__ gamma,
                       const float* __restrict__ beta, const float* __restrict__ mmean,
                       const float* __restrict__ mvar, const float* __restrict__ Wd,
                       const float* __restrict__ bd, const float* __restrict__ y,
                       float* __restrict__ logits_out, float* __restrict__ probs_out,
                       float* __restrict__ rloss)
{
  constexpr int H = 512, C = 100;
  __shared__ float xn[H];
  __shared__ float red[128];
  const int b = blockIdx.x, tid = threadIdx.x;

  for (int k = tid; k < H; k += 128) {
    const float h = hlast[(size_t)b * H + k];
    xn[k] = (h - mmean[k]) * rsqrtf(mvar[k] + 1e-3f) * gamma[k] + beta[k];
  }
  __syncthreads();

  float logit = 0.0f;
  if (tid < C) {
    float acc = bd[tid];
    for (int k = 0; k < H; ++k) acc = fmaf(xn[k], Wd[k * C + tid], acc);
    logit = acc;
  }

  red[tid] = (tid < C) ? logit : -3.4e38f;
  __syncthreads();
  for (int s = 64; s > 0; s >>= 1) { if (tid < s) red[tid] = fmaxf(red[tid], red[tid + s]); __syncthreads(); }
  const float mx = red[0];
  __syncthreads();

  const float e = (tid < C) ? __expf(logit - mx) : 0.0f;
  red[tid] = e;
  __syncthreads();
  for (int s = 64; s > 0; s >>= 1) { if (tid < s) red[tid] += red[tid + s]; __syncthreads(); }
  const float sum = red[0];
  __syncthreads();

  float l = 0.0f;
  if (tid < C) {
    logits_out[b * C + tid] = logit;
    probs_out[b * C + tid]  = e / sum;
    l = -y[b * C + tid] * ((logit - mx) - logf(sum));
  }
  red[tid] = l;
  __syncthreads();
  for (int s = 64; s > 0; s >>= 1) { if (tid < s) red[tid] += red[tid + s]; __syncthreads(); }
  if (tid == 0) rloss[b] = red[0];
}

__global__ void k_reduce(const float* __restrict__ rloss, float* __restrict__ loss) {
  __shared__ float red[128];
  const int tid = threadIdx.x;
  red[tid] = rloss[tid];
  __syncthreads();
  for (int s = 64; s > 0; s >>= 1) { if (tid < s) red[tid] += red[tid + s]; __syncthreads(); }
  if (tid == 0) *loss = red[0] * (1.0f / 128.0f);
}

// ---------------- launch ----------------
extern "C" void kernel_launch(void* const* d_in, const int* in_sizes, int n_in,
                              void* d_out, int out_size, void* d_ws, size_t ws_size,
                              hipStream_t stream)
{
  (void)in_sizes; (void)n_in; (void)out_size; (void)ws_size;
  const float* x     = (const float*)d_in[0];
  const float* y     = (const float*)d_in[1];
  const float* Wx    = (const float*)d_in[2];
  const float* Wh    = (const float*)d_in[3];
  const float* b     = (const float*)d_in[4];
  const float* gamma = (const float*)d_in[5];
  const float* beta  = (const float*)d_in[6];
  const float* mmean = (const float*)d_in[7];
  const float* mvar  = (const float*)d_in[8];
  const float* Wd    = (const float*)d_in[9];
  const float* bd    = (const float*)d_in[10];
  float* out = (float*)d_out;

  char* ws = (char*)d_ws;                                   // ~19.9 MB used
  unsigned short* Wxb  = (unsigned short*)(ws + 0);         // 524288 B
  unsigned short* Whb  = (unsigned short*)(ws + 524288);    // 2097152 B
  unsigned short* xbf  = (unsigned short*)(ws + 2621440);   // 16777216 B
  unsigned short* hbuf = (unsigned short*)(ws + 19398656);  // 262144 B (2 buffers)
  float*          hl   = (float*)(ws + 19660800);           // 262144 B
  float*          rl   = (float*)(ws + 19922944);           // 512 B
  unsigned*       bar  = (unsigned*)(ws + 19923456);        // 4 B

  k_f32_to_bf16<<<(262144 + 255) / 256, 256, 0, stream>>>(Wx, Wxb, 262144);
  k_f32_to_bf16<<<(1048576 + 255) / 256, 256, 0, stream>>>(Wh, Whb, 1048576);
  k_f32_to_bf16<<<(8388608 + 255) / 256, 256, 0, stream>>>(x, xbf, 8388608);
  k_init<<<(131072 + 255) / 256, 256, 0, stream>>>(hbuf, bar, 131072);

  k_lstm<<<32, 256, 0, stream>>>(Wxb, Whb, xbf, b, hbuf, hl, bar);

  k_head<<<128, 128, 0, stream>>>(hl, gamma, beta, mmean, mvar, Wd, bd, y,
                                  out, out + 12800, rl);
  k_reduce<<<1, 128, 0, stream>>>(rl, out + 25600);
}
